// BEVFormerLite_55619826483469
// MI455X (gfx1250) — matrix operand order repaired and checked by reading, not compiled, not run on hardware
//
#include <hip/hip_runtime.h>

typedef __attribute__((ext_vector_type(16))) _Float16 v16h;
typedef __attribute__((ext_vector_type(8)))  _Float16 v8h;
typedef __attribute__((ext_vector_type(8)))  float    v8f;

#define B_    2
#define NCAM  6
#define CCH   256
#define FH    28
#define FW    50
#define NPIX  (FH*FW)          // 1400
#define NPTS  40000            // 200*200
#define NZ    3
#define PTILE 64               // points per GEMM block

// ---------------------------------------------------------------------------
// Kernel 0: transpose feats (B,N,C,H,W) f32 -> featsT (B,N,H*W,C) f16.
// Channel-innermost layout makes the per-corner 256-channel gather in the
// sampling kernel a single coalesced 512B read (feats is L2-resident: 17MB).
// ---------------------------------------------------------------------------
__global__ __launch_bounds__(256) void k_transpose_feats(
    const float* __restrict__ feats, _Float16* __restrict__ featsT)
{
  int o    = blockIdx.x * 256 + threadIdx.x;   // linear over output
  int c    = o & 255;
  int rest = o >> 8;                           // bn*1400 + yx  (max 16800)
  int yx   = rest % NPIX;
  int bn   = rest / NPIX;
  featsT[o] = (_Float16)feats[((size_t)bn * CCH + c) * NPIX + yx];
}

// ---------------------------------------------------------------------------
// Kernel 0b: conv_w f32 -> f16, and fold BN+bias into per-o scale/shift:
//   y = relu( gemm*scale[o] + shift[o] )
// ---------------------------------------------------------------------------
__global__ __launch_bounds__(256) void k_prep_params(
    const float* __restrict__ conv_w, const float* __restrict__ conv_b,
    const float* __restrict__ gamma,  const float* __restrict__ beta,
    const float* __restrict__ mean,   const float* __restrict__ var,
    _Float16* __restrict__ Wf16, float* __restrict__ fscale, float* __restrict__ fshift)
{
  int t = blockIdx.x * 256 + threadIdx.x;       // 0..65535
  Wf16[t] = (_Float16)conv_w[t];
  if (blockIdx.x == 0) {
    float inv  = gamma[t] * rsqrtf(var[t] + 1e-5f);
    fscale[t] = inv;
    fshift[t] = beta[t] + (conv_b[t] - mean[t]) * inv;
  }
}

// ---------------------------------------------------------------------------
// Kernel 1: projection + bilinear sample + cam/z fusion.
// Block = 32 BEV points, 256 threads (lane = channel -> coalesced gathers,
// wave-uniform validity branches). Writes bev as f16 [b][c][p] via an LDS
// transpose tile so the global store is coalesced along p.
// ---------------------------------------------------------------------------
__global__ __launch_bounds__(256) void k_bev_sample(
    const _Float16* __restrict__ featsT,   // [b][n][yx][c] f16
    const float*    __restrict__ intr,     // [b][n][3][3]
    const float*    __restrict__ extr,     // [b][n][4][4]
    const float*    __restrict__ bev_pts,  // [3][40000][3]
    _Float16*       __restrict__ bevf16)   // [b][c][40000]
{
  __shared__ float sx[32][NZ*NCAM];
  __shared__ float sy[32][NZ*NCAM];
  __shared__ float sv[32][NZ*NCAM];
  __shared__ _Float16 tile[32][CCH + 8];   // +8 pad: kill bank conflicts in phase 3

  const int t  = threadIdx.x;
  const int b  = blockIdx.y;
  const int p0 = blockIdx.x * 32;

  // ---- phase 1: 32 pts x 3 z x 6 cams = 576 projections, cooperatively ----
  for (int idx = t; idx < 32 * NZ * NCAM; idx += 256) {
    int pt = idx / (NZ * NCAM);
    int cb = idx - pt * (NZ * NCAM);
    int z  = cb / NCAM;
    int n  = cb - z * NCAM;
    const float* E = extr + (size_t)(b * NCAM + n) * 16;
    const float* K = intr + (size_t)(b * NCAM + n) * 9;
    const float* w = bev_pts + ((size_t)z * NPTS + p0 + pt) * 3;
    // E is rigid: E_inv = [R^T, -R^T t]  ->  cam_i = sum_j R[j][i]*(w_j - t_j)
    float wx = w[0] - E[3], wy = w[1] - E[7], wz = w[2] - E[11];
    float cx = E[0]*wx + E[4]*wy + E[8] *wz;
    float cy = E[1]*wx + E[5]*wy + E[9] *wz;
    float cz = E[2]*wx + E[6]*wy + E[10]*wz;
    float pzr = 1.0f / (cz + 1e-6f);              // K row2 = [0,0,1] -> p2 = cz
    float u = (K[0]*cx + K[1]*cy + K[2]*cz) * pzr;
    float v = (K[3]*cx + K[4]*cy + K[5]*cz) * pzr;
    float un = u * ((float)FW / 1600.0f) / (FW - 1.0f) * 2.0f - 1.0f;
    float vn = v * ((float)FH /  900.0f) / (FH - 1.0f) * 2.0f - 1.0f;
    bool ok = (cz > 0.1f) & (un >= -1.f) & (un <= 1.f) & (vn >= -1.f) & (vn <= 1.f);
    sx[pt][cb] = (un + 1.f) * 0.5f * (float)(FW - 1);
    sy[pt][cb] = (vn + 1.f) * 0.5f * (float)(FH - 1);
    sv[pt][cb] = ok ? 1.f : 0.f;
  }
  __syncthreads();

  // ---- phase 2: lane = channel; gather & fuse over cams and z ----
  const int c = t;
  for (int pt = 0; pt < 32; ++pt) {
    float acc[NZ] = {0.f, 0.f, 0.f};
    float cnt[NZ] = {0.f, 0.f, 0.f};
    #pragma unroll
    for (int z = 0; z < NZ; ++z) {
      for (int n = 0; n < NCAM; ++n) {
        int cb = z * NCAM + n;
        if (sv[pt][cb] > 0.f) {                       // wave-uniform branch
          float x = sx[pt][cb], y = sy[pt][cb];
          float x0f = floorf(x), y0f = floorf(y);
          int   x0 = (int)x0f,  y0 = (int)y0f;
          float wx1 = x - x0f,  wy1 = y - y0f;
          float wx0 = 1.f - wx1, wy0 = 1.f - wy1;
          const _Float16* fb = featsT + ((size_t)(b*NCAM + n) * NPIX) * CCH + c;
          float s = 0.f;
          #pragma unroll
          for (int dy = 0; dy < 2; ++dy) {
            #pragma unroll
            for (int dx = 0; dx < 2; ++dx) {
              int xi = x0 + dx, yi = y0 + dy;
              float ww = (dx ? wx1 : wx0) * (dy ? wy1 : wy0);
              if (xi >= 0 && xi < FW && yi >= 0 && yi < FH)
                s += (float)fb[(size_t)(yi * FW + xi) * CCH] * ww;
            }
          }
          acc[z] += s;
          cnt[z] += 1.f;
        }
      }
    }
    float bev = 0.f;
    #pragma unroll
    for (int z = 0; z < NZ; ++z) bev += acc[z] / (cnt[z] + 1e-6f);
    tile[pt][c] = (_Float16)(bev * (1.0f / 3.0f));
  }
  __syncthreads();

  // ---- phase 3: coalesced f16 write-out, lanes = points ----
  {
    int m  = t & 31;
    int c0 = (t >> 5) * 32;
    for (int i = 0; i < 32; ++i) {
      int cc = c0 + i;
      bevf16[((size_t)b * CCH + cc) * NPTS + p0 + m] = tile[m][cc];
    }
  }
}

// ---------------------------------------------------------------------------
// Kernel 2: fused 1x1-conv GEMM + BN + ReLU via v_wmma_f32_16x16x32_f16.
// Block = 256 thr (8 waves) computes 256 o x 64 p. Wave w owns o-half
// (w>>2)*128 (8 o-tiles) and p-tile (w&3)*16 -> 8 accumulators, 8 WMMA/K-step.
// Double-buffered LDS; the 16KB W slice is staged with
// GLOBAL_LOAD_ASYNC_TO_LDS_B128 (ASYNCcnt) overlapping the WMMA work.
// Fragment layouts used here (wave32; contiguous K in A/B, not ISA table order):
//   A f16 16x32 : lane&15 = M row; elem e -> K = 16*(lane>=16)+e
//   B f16 32x16 : lane&15 = N col; elem e -> K = 16*(lane>=16)+e
//   D f32 16x16 : lane&15 = N;     reg  r -> M = r + 8*(lane>=16)
// ---------------------------------------------------------------------------
__global__ __launch_bounds__(256) void k_conv_wmma(
    const _Float16* __restrict__ Wf16,     // [256][256]
    const _Float16* __restrict__ bevf16,   // [b][256][40000]
    const float*    __restrict__ fscale,
    const float*    __restrict__ fshift,
    float*          __restrict__ out)      // [b][256][40000]
{
  __shared__ __align__(64) _Float16 Wlds[2][CCH * 32];    // 2 x 16 KB  [o][kk]
  __shared__ __align__(64) _Float16 Bt  [2][PTILE * 32];  // 2 x  4 KB  [p][kk]

  const int t    = threadIdx.x;
  const int lane = t & 31;
  const int wave = t >> 5;
  const int b    = blockIdx.y;
  const int p0   = blockIdx.x * PTILE;
  const int m    = lane & 15;
  const int mh   = (lane >= 16) ? 1 : 0;
  const int klo  = mh * 8;                  // A-frag K base is 2*klo = 16*mh
  const int og   = wave >> 2;               // o-half: 0 or 1 (128 outputs each)
  const int ptb  = (wave & 3) * 16;         // p-tile base within block
  const _Float16* bevb = bevf16 + (size_t)b * CCH * NPTS;

  v8f acc[8];
  #pragma unroll
  for (int j = 0; j < 8; ++j) acc[j] = (v8f){};

  // async stage of the 16KB W slice: 1024 x 16B lane-copies, 4 per thread
  auto stage_W = [&](int k0, int buf) {
    #pragma unroll
    for (int r = 0; r < 4; ++r) {
      int chunk = t + r * 256;              // 0..1023
      int o     = chunk >> 2;
      int part  = chunk & 3;                // 4 x 16B per 64B row
      const _Float16* g = Wf16 + (size_t)o * CCH + k0 + part * 8;
      unsigned lds = (unsigned)(size_t)(&Wlds[buf][o * 32 + part * 8]);
      asm volatile("global_load_async_to_lds_b128 %0, %1, off"
                   :: "v"(lds), "v"(g) : "memory");
    }
  };
  // synchronous transposed stage of the 4KB bev tile
  auto stage_B = [&](int k0, int buf) {
    int kk = t & 31;
    int pg = t >> 5;                        // 0..7
    v8h val = *(const v8h*)(bevb + (size_t)(k0 + kk) * NPTS + p0 + pg * 8);
    #pragma unroll
    for (int i = 0; i < 8; ++i)
      Bt[buf][(pg * 8 + i) * 32 + kk] = val[i];
  };

  stage_W(0, 0);
  stage_B(0, 0);
  asm volatile("s_wait_asynccnt 0x0" ::: "memory");
  __syncthreads();

  for (int ks = 0; ks < CCH / 32; ++ks) {
    const int cur = ks & 1;
    if (ks + 1 < CCH / 32) {                // prefetch next slice into other buffer
      stage_W((ks + 1) * 32, cur ^ 1);
      stage_B((ks + 1) * 32, cur ^ 1);
    }

    // B fragment for this wave's p-tile: one aligned 32B v16h read
    v16h fb = *(const v16h*)(&Bt[cur][(ptb + m) * 32 + 16 * mh]);

    #pragma unroll
    for (int j = 0; j < 8; ++j) {
      const _Float16* wr = &Wlds[cur][(og * 128 + j * 16 + m) * 32];
      v8h alo = *(const v8h*)(wr + 2 * klo);
      v8h ahi = *(const v8h*)(wr + 2 * klo + 8);
      v16h fa;
      #pragma unroll
      for (int e = 0; e < 8; ++e) { fa[e] = alo[e]; fa[8 + e] = ahi[e]; }
      acc[j] = __builtin_amdgcn_wmma_f32_16x16x32_f16(
                   false, fa, false, fb, (short)0, acc[j], false, false);
    }

    asm volatile("s_wait_asynccnt 0x0" ::: "memory");  // next W slice landed
    __syncthreads();                                   // + Bt stores / cur reads done
  }

  // epilogue: fused scale/shift (bias+BN) + ReLU; lanes 0-15 = 16 consecutive p
  float* outb = out + (size_t)b * CCH * NPTS;
  const int p = p0 + ptb + m;
  #pragma unroll
  for (int j = 0; j < 8; ++j) {
    int obase = og * 128 + j * 16 + mh * 8;
    #pragma unroll
    for (int r = 0; r < 8; ++r) {
      int o = obase + r;
      float val = acc[j][r] * fscale[o] + fshift[o];
      outb[(size_t)o * NPTS + p] = fmaxf(val, 0.f);
    }
  }
}

// ---------------------------------------------------------------------------
extern "C" void kernel_launch(void* const* d_in, const int* in_sizes, int n_in,
                              void* d_out, int out_size, void* d_ws, size_t ws_size,
                              hipStream_t stream)
{
  (void)in_sizes; (void)n_in; (void)out_size; (void)ws_size;
  const float* feats   = (const float*)d_in[0];
  const float* intr    = (const float*)d_in[1];
  const float* extr    = (const float*)d_in[2];
  const float* bev_pts = (const float*)d_in[3];
  const float* conv_w  = (const float*)d_in[4];
  const float* conv_b  = (const float*)d_in[5];
  const float* gamma   = (const float*)d_in[6];
  const float* beta    = (const float*)d_in[7];
  const float* mean    = (const float*)d_in[8];
  const float* var     = (const float*)d_in[9];
  float* out = (float*)d_out;

  // workspace layout (all offsets 256B-aligned)
  char* ws = (char*)d_ws;
  _Float16* featsT = (_Float16*)ws;                              //  8,601,600 B
  _Float16* bevf16 = (_Float16*)(ws + 8601600);                  // 40,960,000 B
  _Float16* Wf16   = (_Float16*)(ws + 8601600 + 40960000);       //    131,072 B
  float*    fscale = (float*)   (ws + 8601600 + 40960000 + 131072);
  float*    fshift = fscale + CCH;

  // 0) transpose feats to channel-innermost f16 (4,300,800 elems / 256)
  k_transpose_feats<<<dim3((B_*NCAM*NPIX*CCH)/256), dim3(256), 0, stream>>>(feats, featsT);

  // 0b) weights -> f16, fold BN+bias
  k_prep_params<<<dim3(CCH), dim3(256), 0, stream>>>(
      conv_w, conv_b, gamma, beta, mean, var, Wf16, fscale, fshift);

  // 1) projection + bilinear sampling + fusion -> bev f16 [b][c][p]
  k_bev_sample<<<dim3(NPTS/32, B_), dim3(256), 0, stream>>>(
      featsT, intr, extr, bev_pts, bevf16);

  // 2) fused 1x1-conv GEMM (WMMA, async double-buffered) + BN + ReLU
  k_conv_wmma<<<dim3(NPTS/PTILE, B_), dim3(256), 0, stream>>>(
      Wf16, bevf16, fscale, fshift, out);
}
